// MultiHeadAttention_80453327389411
// MI455X (gfx1250) — matrix operand order repaired and checked by reading, not compiled, not run on hardware
//
#include <hip/hip_runtime.h>

// ---------------------------------------------------------------------------
// MultiHeadAttention + exact 1.5-entmax for MI455X (gfx1250, wave32, WMMA)
//
// Pipeline:
//   1) proj_bf16_kernel   : P = X @ W^T + b  (f32 in, bf16 out) -- v_wmma bf16,
//                           16x32 output tile per wave (A-fragment reuse x2)
//   2) scores_wmma_kernel : S[b,q,k] = (Qp . Kp)/8 per head     -- v_wmma bf16,
//                           16x64 output tile per wave (A-fragment reuse x4)
//   3) entmax_kernel      : sort-based 1.5-entmax per row, += p/H into out
// Heads are looped on the host (12 rounds of kernels 2+3) so the score
// workspace is one head (18.9 MB); total ws = 28.3 MB.
// ---------------------------------------------------------------------------

typedef __attribute__((ext_vector_type(16))) __bf16 v16bf;
typedef __attribute__((ext_vector_type(8)))  __bf16 v8bf;
typedef __attribute__((ext_vector_type(8)))  float  v8f;

static __device__ __forceinline__ __bf16 f2bf(float x) {
  // native convert -> v_cvt_*bf16_f32 (RNE), avoids manual bit-twiddling VALU
  return (__bf16)x;
}

// ---------------------------------------------------------------------------
// Kernel 1: P[M,N] = X[M,K] @ W[N,K]^T + bias[N], output bf16.
// 4 waves/block; wave w owns a 16(M)x32(N) tile: one A fragment feeds two
// WMMAs per K-step (K-step = 32).
// ---------------------------------------------------------------------------
__global__ void proj_bf16_kernel(const float* __restrict__ X,
                                 const float* __restrict__ W,
                                 const float* __restrict__ bias,
                                 __bf16* __restrict__ P,
                                 int M, int N, int Kd) {
  const int lane = threadIdx.x & 31;
  const int wave = threadIdx.x >> 5;
  const int m0 = blockIdx.x * 64 + wave * 16;
  const int n0 = blockIdx.y * 32;
  const int r  = lane & 15;   // row within fragment
  const int hi = lane >> 4;   // half-wave (K-chunk selector)

  const float* arow  = X + (size_t)(m0 + r) * Kd;        // A: contiguous K
  const float* b0row = W + (size_t)(n0 + r) * Kd;        // B^T tile 0
  const float* b1row = W + (size_t)(n0 + 16 + r) * Kd;   // B^T tile 1

  v8f acc0 = {}, acc1 = {};
  for (int kk = 0; kk < Kd; kk += 32) {
    // A fragment: 16 contiguous K values [base..+15], base = kk + hi*16
    const float4* ap = (const float4*)(arow + kk + hi * 16);
    float4 al0 = ap[0], al1 = ap[1];   // k  base..base+7
    float4 ah0 = ap[2], ah1 = ap[3];   // k  base+8..base+15
    float av[16] = {al0.x, al0.y, al0.z, al0.w, al1.x, al1.y, al1.z, al1.w,
                    ah0.x, ah0.y, ah0.z, ah0.w, ah1.x, ah1.y, ah1.z, ah1.w};
    v16bf a;
#pragma unroll
    for (int t = 0; t < 16; ++t) a[t] = f2bf(av[t]);

    // B fragments: 16 contiguous K values at kk + hi*16
    const float4* bp0 = (const float4*)(b0row + kk + hi * 16);
    const float4* bp1 = (const float4*)(b1row + kk + hi * 16);
    float4 c0 = bp0[0], c1 = bp0[1], c2 = bp0[2], c3 = bp0[3];
    float4 d0 = bp1[0], d1 = bp1[1], d2 = bp1[2], d3 = bp1[3];
    float bv0[16] = {c0.x, c0.y, c0.z, c0.w, c1.x, c1.y, c1.z, c1.w,
                     c2.x, c2.y, c2.z, c2.w, c3.x, c3.y, c3.z, c3.w};
    float bv1[16] = {d0.x, d0.y, d0.z, d0.w, d1.x, d1.y, d1.z, d1.w,
                     d2.x, d2.y, d2.z, d2.w, d3.x, d3.y, d3.z, d3.w};
    v16bf bb0, bb1;
#pragma unroll
    for (int t = 0; t < 16; ++t) { bb0[t] = f2bf(bv0[t]); bb1[t] = f2bf(bv1[t]); }

    acc0 = __builtin_amdgcn_wmma_f32_16x16x32_bf16(false, a, false, bb0,
                                                   (short)0, acc0, false, false);
    acc1 = __builtin_amdgcn_wmma_f32_16x16x32_bf16(false, a, false, bb1,
                                                   (short)0, acc1, false, false);
  }

  const float bias0 = bias[n0 + r];
  const float bias1 = bias[n0 + 16 + r];
  const int mbase = m0 + hi * 8;            // C/D layout: M = v + 8*hi, N = r
#pragma unroll
  for (int v = 0; v < 8; ++v) {
    P[(size_t)(mbase + v) * N + (n0 + r)]      = f2bf(acc0[v] + bias0);
    P[(size_t)(mbase + v) * N + (n0 + 16 + r)] = f2bf(acc1[v] + bias1);
  }
}

// ---------------------------------------------------------------------------
// Kernel 2: per-head scores S[b,q,k] = (1/8) * sum_d Qp[b,q,h*64+d]*Kp[b,k,h*64+d]
// Wave owns a 16(q) x 64(k) tile: per K-step one A fragment feeds 4 WMMAs.
// Operands are bf16 in the workspace already (no conversion).
// ---------------------------------------------------------------------------
__global__ void scores_wmma_kernel(const __bf16* __restrict__ Qp,
                                   const __bf16* __restrict__ Kp,
                                   float* __restrict__ S, int h) {
  const int L = 1536, D = 768;
  const int lane = threadIdx.x & 31;
  const int wave = threadIdx.x >> 5;
  const int b  = blockIdx.z;
  const int q0 = blockIdx.x * 64 + wave * 16;
  const int k0 = blockIdx.y * 64;
  const int r  = lane & 15;
  const int hi = lane >> 4;

  const __bf16* arow = Qp + (size_t)(b * L + q0 + r) * D + h * 64;
  const __bf16* kbase = Kp + (size_t)(b * L) * D + h * 64;

  v8f acc[4] = {{}, {}, {}, {}};
#pragma unroll
  for (int kk = 0; kk < 64; kk += 32) {
    const v8bf* ap = (const v8bf*)(arow + kk + hi * 16); // 16B-aligned
    v8bf lo8 = ap[0];
    v8bf hi8 = ap[1];                                    // +8 elements
    v16bf a;
#pragma unroll
    for (int t = 0; t < 8; ++t) { a[t] = lo8[t]; a[t + 8] = hi8[t]; }
#pragma unroll
    for (int n = 0; n < 4; ++n) {
      v16bf bb = *(const v16bf*)(kbase + (size_t)(k0 + n * 16 + r) * D
                                 + kk + hi * 16);        // 32B-aligned
      acc[n] = __builtin_amdgcn_wmma_f32_16x16x32_bf16(false, a, false, bb,
                                                       (short)0, acc[n],
                                                       false, false);
    }
  }

  float* sbase = S + (size_t)b * L * L;
  const int qbase = q0 + hi * 8;
#pragma unroll
  for (int n = 0; n < 4; ++n)
#pragma unroll
    for (int v = 0; v < 8; ++v)
      sbase[(size_t)(qbase + v) * L + (k0 + n * 16 + r)] =
          acc[n][v] * 0.125f;                            // 1/sqrt(64)
}

// ---------------------------------------------------------------------------
// Kernel 3: exact sort-based 1.5-entmax over one (b,q) row (length 1536),
// masked keys -> -1e9 first. Accumulates p/12 into out (write on head 0).
// One 256-thread block per row; everything LDS-resident (~31 KB).
// ---------------------------------------------------------------------------
#define LK 1536
#define NS 2048   // padded pow2 for bitonic sort / scans

__global__ void entmax_kernel(const float* __restrict__ S,
                              const int* __restrict__ mask,
                              float* __restrict__ out, int accumulate) {
  __shared__ float zorig[LK];
  __shared__ float zs[NS];
  __shared__ float cs[NS];
  __shared__ float css[NS];
  __shared__ float red[256];
  __shared__ int scount;

  const int tid = threadIdx.x;
  const int q = blockIdx.x;
  const int b = blockIdx.y;
  const float* srow = S + ((size_t)b * LK + q) * LK;
  const int* mrow = mask + (size_t)b * LK;

  // z = s/2 (masked -> -1e9/2), find max
  float lmax = -3.4e38f;
  for (int i = tid; i < LK; i += 256) {
    float s = (mrow[i] != 0) ? srow[i] : -1.0e9f;
    float z = 0.5f * s;
    zorig[i] = z;
    lmax = fmaxf(lmax, z);
  }
  red[tid] = lmax;
  __syncthreads();
  for (int off = 128; off > 0; off >>= 1) {
    if (tid < off) red[tid] = fmaxf(red[tid], red[tid + off]);
    __syncthreads();
  }
  const float zmax = red[0];
  __syncthreads();

  // shift and fill sort buffer (pad with -FLT_MAX so padding sinks to the tail)
  for (int i = tid; i < NS; i += 256) {
    if (i < LK) { float z = zorig[i] - zmax; zorig[i] = z; zs[i] = z; }
    else        { zs[i] = -3.4e38f; }
  }
  __syncthreads();

  // bitonic sort, descending
  for (int k = 2; k <= NS; k <<= 1) {
    for (int j = k >> 1; j > 0; j >>= 1) {
      for (int t = tid; t < NS; t += 256) {
        int ixj = t ^ j;
        if (ixj > t) {
          float x = zs[t], y = zs[ixj];
          bool up = ((t & k) == 0);
          if (up ? (x < y) : (x > y)) { zs[t] = y; zs[ixj] = x; }
        }
      }
      __syncthreads();
    }
  }

  // inclusive scans of zs and zs^2 over the first LK entries (pad = 0)
  for (int i = tid; i < NS; i += 256) {
    float v = (i < LK) ? zs[i] : 0.0f;
    cs[i] = v; css[i] = v * v;
  }
  __syncthreads();
  for (int off = 1; off < NS; off <<= 1) {
    float a0[8], b0[8], a1[8], b1[8];
#pragma unroll
    for (int t = 0; t < 8; ++t) {
      int i = tid + t * 256;
      a0[t] = cs[i];  b0[t] = (i >= off) ? cs[i - off]  : 0.0f;
      a1[t] = css[i]; b1[t] = (i >= off) ? css[i - off] : 0.0f;
    }
    __syncthreads();
#pragma unroll
    for (int t = 0; t < 8; ++t) {
      int i = tid + t * 256;
      cs[i]  = a0[t] + b0[t];
      css[i] = a1[t] + b1[t];
    }
    __syncthreads();
  }

  // tau_j and support size (reference math, f32 throughout)
  if (tid == 0) scount = 0;
  __syncthreads();
  int cnt = 0;
  for (int i = tid; i < LK; i += 256) {
    float kf = (float)(i + 1);
    float mean = cs[i] / kf;
    float msq  = css[i] / kf;
    float ssv  = kf * (msq - mean * mean);
    float delta = (1.0f - ssv) / kf;
    float tau = mean - sqrtf(fmaxf(delta, 0.0f));
    css[i] = tau;                 // overwrite in place (thread-private index)
    if (tau <= zs[i]) cnt++;
  }
  atomicAdd(&scount, cnt);
  __syncthreads();
  const int support = scount;     // >= 1 always (tau_1 = z_(1) - 1)
  const float tau_star = css[support - 1];

  // p = clip(z - tau*, 0)^2 ; out += p / H
  const float inv_h = 1.0f / 12.0f;
  float* orow = out + ((size_t)b * LK + q) * LK;
  for (int i = tid; i < LK; i += 256) {
    float d = zorig[i] - tau_star;
    float p = fmaxf(d, 0.0f);
    float val = p * p * inv_h;
    orow[i] = accumulate ? (orow[i] + val) : val;
  }
}

// ---------------------------------------------------------------------------
extern "C" void kernel_launch(void* const* d_in, const int* in_sizes, int n_in,
                              void* d_out, int out_size, void* d_ws, size_t ws_size,
                              hipStream_t stream) {
  (void)in_sizes; (void)n_in; (void)out_size; (void)ws_size;
  const float* query = (const float*)d_in[0];   // [2,1536,768] f32
  const float* key   = (const float*)d_in[1];   // [2,1536,768] f32
  const int*   mask  = (const int*)  d_in[2];   // [2,1536] i32
  const float* wq_w  = (const float*)d_in[3];   // [768,768]
  const float* wq_b  = (const float*)d_in[4];   // [768]
  const float* wk_w  = (const float*)d_in[5];   // [768,768]
  const float* wk_b  = (const float*)d_in[6];   // [768]
  float* out = (float*)d_out;                   // [2,1536,1536] f32

  const int B = 2, L = 1536, D = 768, H = 12;

  // workspace: Qp bf16 (4.72MB) | Kp bf16 (4.72MB) | S f32 one head (18.87MB)
  __bf16* Qp = (__bf16*)d_ws;
  __bf16* Kp = Qp + (size_t)B * L * D;
  float*  S  = (float*)((char*)d_ws + 2u * (size_t)B * L * D * sizeof(__bf16));

  // projections: (B*L=3072) rows x (D=768) cols; block = 64(M) x 32(N)
  dim3 pgrid((B * L) / 64, D / 32);             // (48, 24)
  proj_bf16_kernel<<<pgrid, 128, 0, stream>>>(query, wq_w, wq_b, Qp, B * L, D, D);
  proj_bf16_kernel<<<pgrid, 128, 0, stream>>>(key,   wk_w, wk_b, Kp, B * L, D, D);

  dim3 sgrid(L / 64, L / 64, B);                // (24, 24, 2); block = 64q x 64k
  dim3 egrid(L, B);                             // one block per (q, b) row
  for (int h = 0; h < H; ++h) {
    scores_wmma_kernel<<<sgrid, 128, 0, stream>>>(Qp, Kp, S, h);
    entmax_kernel<<<egrid, 256, 0, stream>>>(S, mask, out, (h == 0) ? 0 : 1);
  }
}
